// GAT_19499151524576
// MI455X (gfx1250) — matrix operand order repaired and checked by reading, not compiled, not run on hardware
//
#include <hip/hip_runtime.h>
#include <hip/hip_bf16.h>

// ---- problem constants (match reference) ----
#define NIN   128
#define NHEAD 8
#define DHEAD 16
#define QKVC  384          // q|k|v concatenated columns
#define DFF   512
#define INV_SQRT128 0.08838834764831845f

typedef _Float16 v16h __attribute__((ext_vector_type(16)));
typedef _Float16 v8h  __attribute__((ext_vector_type(8)));
typedef float    v8f  __attribute__((ext_vector_type(8)));

typedef __attribute__((address_space(3))) _Float16 lds_h;

// ---------------- WMMA helpers ----------------
__device__ __forceinline__ v8f wmma_f16(v16h a, v16h b, v8f c) {
    // D = A(16x32 f16) * B(32x16 f16) + C(16x16 f32)
    return __builtin_amdgcn_wmma_f32_16x16x32_f16(
        /*neg_a=*/false, a, /*neg_b=*/false, b,
        /*c_mod=*/(short)0, c, /*reuse_a=*/false, /*reuse_b=*/false);
}

// A fragment 16x32 from row-major tile (row 0 at p), leading dim lda (elements).
// Contiguous K order (matches load_frag_b): lanes 0-15 -> M=lane, K=koff..koff+15; koff=(lane>>4)*16
__device__ __forceinline__ v16h load_frag_a(const _Float16* p, int lda, int kb, int lane) {
    int m    = lane & 15;
    int koff = (lane >> 4) * 16;
    const _Float16* r = p + m * lda + kb + koff;
    v8h lo = *(const v8h*)(r);        // K = kb+koff .. +7      -> VGPR 0-3
    v8h hi = *(const v8h*)(r + 8);    // K = kb+koff+8 .. +15   -> VGPR 4-7
    v16h o;
#pragma unroll
    for (int i = 0; i < 8; ++i) { o[i] = lo[i]; o[i + 8] = hi[i]; }
    return o;
}

// B fragment 32x16 from N-major weights: pT[n*ldk + k] (K contiguous per column).
// lanes 0-15 -> N=lane, K=kb..kb+15 ; lanes 16-31 -> N=lane-16, K=kb+16..kb+31
__device__ __forceinline__ v16h load_frag_b(const _Float16* pT, int ldk, int kb, int lane) {
    int n    = lane & 15;
    int koff = (lane >> 4) * 16;
    return *(const v16h*)(pT + n * ldk + kb + koff);
}

__device__ __forceinline__ float wave_sum(float v) {
#pragma unroll
    for (int off = 16; off > 0; off >>= 1) v += __shfl_xor(v, off, 32);
    return v;
}

// order-preserving float<->uint mapping for atomicMax on floats
__device__ __forceinline__ unsigned fkey(float f) {
    unsigned u = __float_as_uint(f);
    return (u & 0x80000000u) ? ~u : (u | 0x80000000u);
}
__device__ __forceinline__ float funkey(unsigned u) {
    unsigned v = (u & 0x80000000u) ? (u & 0x7fffffffu) : ~u;
    return __uint_as_float(v);
}

// ---------------- prep kernels ----------------
__global__ void k_f32_to_f16(const float* __restrict__ x, _Float16* __restrict__ y, int n) {
    int i = blockIdx.x * blockDim.x + threadIdx.x;
    if (i < n) y[i] = (_Float16)x[i];
}

__global__ void k_build_wqkvT(const float* __restrict__ Wq, const float* __restrict__ Wk,
                              const float* __restrict__ Wv, _Float16* __restrict__ wT) {
    int i = blockIdx.x * blockDim.x + threadIdx.x;      // over 384*128
    if (i >= QKVC * NIN) return;
    int j = i / NIN, kk = i % NIN;                      // j = out col, kk = in feat
    float v;
    if (j < 128)      v = Wq[kk * 128 + j];
    else if (j < 256) v = Wk[kk * 128 + (j - 128)];
    else              v = Wv[kk * 128 + (j - 256)];
    wT[j * NIN + kk] = (_Float16)v;
}

__global__ void k_build_w1T(const float* __restrict__ W1, _Float16* __restrict__ wT) {
    int i = blockIdx.x * blockDim.x + threadIdx.x;      // over 512*128
    if (i >= DFF * NIN) return;
    int j = i / NIN, kk = i % NIN;
    wT[j * NIN + kk] = (_Float16)W1[kk * DFF + j];
}

__global__ void k_build_w2T(const float* __restrict__ W2, _Float16* __restrict__ wT) {
    int i = blockIdx.x * blockDim.x + threadIdx.x;      // over 128*512
    if (i >= NIN * DFF) return;
    int n = i / DFF, k = i % DFF;
    wT[n * DFF + k] = (_Float16)W2[k * NIN + n];
}

__global__ void k_zero(float* __restrict__ p, int n) {
    int i = blockIdx.x * blockDim.x + threadIdx.x;
    if (i < n) p[i] = 0.0f;
}

// ---------------- QKV projection GEMM: qkv16[N,384] = feat16[N,128] @ WqkvT ----------------
__global__ void __launch_bounds__(256) k_qkv_gemm(const _Float16* __restrict__ feat16,
                                                  const _Float16* __restrict__ wT,
                                                  _Float16* __restrict__ qkv16) {
    int wave = threadIdx.x >> 5, lane = threadIdx.x & 31;
    int rbase = blockIdx.x * 16;
    int ct    = blockIdx.y * 8 + wave;     // 0..23 col tiles
    int nb    = ct * 16;
    v8f acc = {};
#pragma unroll
    for (int kb = 0; kb < NIN; kb += 32) {
        v16h a = load_frag_a(feat16 + rbase * NIN, NIN, kb, lane);
        v16h b = load_frag_b(wT + nb * NIN, NIN, kb, lane);
        acc = wmma_f16(a, b, acc);
    }
    int n = lane & 15, mo = (lane >> 4) * 8;
#pragma unroll
    for (int r = 0; r < 8; ++r)
        qkv16[(rbase + mo + r) * QKVC + nb + n] = (_Float16)acc[r];
}

// ---------------- edge kernels ----------------
__global__ void k_edge_score(const _Float16* __restrict__ qkv16,
                             const int* __restrict__ src, const int* __restrict__ dst,
                             float* __restrict__ escore, unsigned* __restrict__ nmax, int E) {
    int t = blockIdx.x * blockDim.x + threadIdx.x;
    if (t >= E * NHEAD) return;
    int e = t >> 3, h = t & 7;
    int s = src[e], d = dst[e];
    v16h kv = *(const v16h*)(qkv16 + (size_t)s * QKVC + 128 + h * DHEAD);
    v16h qv = *(const v16h*)(qkv16 + (size_t)d * QKVC +   0 + h * DHEAD);
    float acc = 0.f;
#pragma unroll
    for (int i = 0; i < DHEAD; ++i) acc += (float)kv[i] * (float)qv[i];
    acc *= INV_SQRT128;
    escore[t] = acc;
    atomicMax(&nmax[d * NHEAD + h], fkey(acc));
}

__global__ void k_edge_exp(const int* __restrict__ dst,
                           float* __restrict__ escore, const unsigned* __restrict__ nmax,
                           float* __restrict__ denom, int E) {
    int t = blockIdx.x * blockDim.x + threadIdx.x;
    if (t >= E * NHEAD) return;
    int e = t >> 3, h = t & 7;
    int d = dst[e];
    float m  = funkey(nmax[d * NHEAD + h]);
    float ex = __expf(escore[t] - m);
    escore[t] = ex;
    atomicAdd(&denom[d * NHEAD + h], ex);
}

__global__ void k_edge_aggregate(const _Float16* __restrict__ qkv16,
                                 const int* __restrict__ src, const int* __restrict__ dst,
                                 const float* __restrict__ escore, const float* __restrict__ denom,
                                 float* __restrict__ ft2, int E) {
    int t = blockIdx.x * blockDim.x + threadIdx.x;
    if (t >= E * NHEAD) return;
    int e = t >> 3, h = t & 7;
    int s = src[e], d = dst[e];
    float a = escore[t] / denom[d * NHEAD + h];
    v16h vv = *(const v16h*)(qkv16 + (size_t)s * QKVC + 256 + h * DHEAD);
    float* dsts = ft2 + (size_t)d * NIN + h * DHEAD;
#pragma unroll
    for (int i = 0; i < DHEAD; ++i) atomicAdd(&dsts[i], a * (float)vv[i]);
}

// ---------------- residual + LayerNorm -> rst32 (f32) & rst16 (f16) ----------------
__global__ void __launch_bounds__(256) k_residual_ln(const float* __restrict__ ft2,
                                                     const float* __restrict__ feat,
                                                     const float* __restrict__ g,
                                                     const float* __restrict__ bb,
                                                     float* __restrict__ rst32,
                                                     _Float16* __restrict__ rst16, int N) {
    int wave = threadIdx.x >> 5, lane = threadIdx.x & 31;
    int node = blockIdx.x * 8 + wave;
    if (node >= N) return;
    const float* f2 = ft2 + (size_t)node * NIN;
    const float* fi = feat + (size_t)node * NIN;
    float x[4], s = 0.f;
#pragma unroll
    for (int j = 0; j < 4; ++j) { x[j] = f2[lane + 32 * j] + fi[lane + 32 * j]; s += x[j]; }
    float mu = wave_sum(s) * (1.0f / NIN);
    float vs = 0.f;
#pragma unroll
    for (int j = 0; j < 4; ++j) { float dd = x[j] - mu; vs += dd * dd; }
    float rs = rsqrtf(wave_sum(vs) * (1.0f / NIN) + 1e-5f);
#pragma unroll
    for (int j = 0; j < 4; ++j) {
        int c = lane + 32 * j;
        float y = (x[j] - mu) * rs * g[c] + bb[c];
        rst32[(size_t)node * NIN + c] = y;
        rst16[(size_t)node * NIN + c] = (_Float16)y;
    }
}

// ---------------- fused FFN (async LDS stage -> GEMM1+PReLU -> GEMM2+residual -> LN) ----------------
__global__ void __launch_bounds__(256) k_ffn_fused(const _Float16* __restrict__ rst16,
                                                   const float* __restrict__ rst32,
                                                   const _Float16* __restrict__ w1T,
                                                   const float* __restrict__ b1,
                                                   const float* __restrict__ pa,
                                                   const _Float16* __restrict__ w2T,
                                                   const float* __restrict__ b2,
                                                   const float* __restrict__ g,
                                                   const float* __restrict__ bb,
                                                   float* __restrict__ out) {
    __shared__ _Float16 sRst[16 * NIN];    // 4 KB
    __shared__ _Float16 sH[16 * DFF];      // 16 KB
    __shared__ float    sOut[16 * NIN];    // 8 KB
    int tid = threadIdx.x, wave = tid >> 5, lane = tid & 31;
    int rbase = blockIdx.x * 16;

    // stage 16x128 rst tile straight into LDS via the CDNA5 async path
    // (GLOBAL_LOAD_ASYNC_TO_LDS_B128, tracked by ASYNCcnt; no VGPR round-trip)
    {
        unsigned ldsAddr = (unsigned)(size_t)((lds_h*)sRst + tid * 8);     // LDS byte address
        const _Float16* gp = rst16 + (size_t)rbase * NIN + tid * 8;        // 16 B per thread
        asm volatile("global_load_async_to_lds_b128 %0, %1, off"
                     :: "v"(ldsAddr), "v"(gp) : "memory");
        asm volatile("s_wait_asynccnt 0x0" ::: "memory");
    }
    __syncthreads();

    // A fragments for K=128, reused across 4 col-tiles per wave
    v16h afrag[4];
#pragma unroll
    for (int k = 0; k < 4; ++k) afrag[k] = load_frag_a(sRst, NIN, k * 32, lane);

    int n = lane & 15, mo = (lane >> 4) * 8;

    // GEMM1: h[16,512] = rst @ W1, + b1, PReLU -> LDS (f16)
#pragma unroll
    for (int cti = 0; cti < 4; ++cti) {
        int nb = (wave * 4 + cti) * 16;
        v8f acc = {};
#pragma unroll
        for (int k = 0; k < 4; ++k) {
            v16h b = load_frag_b(w1T + nb * NIN, NIN, k * 32, lane);
            acc = wmma_f16(afrag[k], b, acc);
        }
        int c = nb + n;
        float bias = b1[c], alpha = pa[c];
#pragma unroll
        for (int r = 0; r < 8; ++r) {
            float h = acc[r] + bias;
            h = (h >= 0.f) ? h : alpha * h;
            sH[(mo + r) * DFF + c] = (_Float16)h;
        }
    }
    __syncthreads();

    // GEMM2: ffn[16,128] = h @ W2, + b2 + rst32 -> LDS (f32)
    {
        int nb = wave * 16;
        v8f acc = {};
        for (int kb = 0; kb < DFF; kb += 32) {
            v16h a = load_frag_a(sH, DFF, kb, lane);
            v16h b = load_frag_b(w2T + nb * DFF, DFF, kb, lane);
            acc = wmma_f16(a, b, acc);
        }
        int c = nb + n;
        float bias = b2[c];
#pragma unroll
        for (int r = 0; r < 8; ++r) {
            int row = mo + r;
            sOut[row * NIN + c] = acc[r] + bias + rst32[(size_t)(rbase + row) * NIN + c];
        }
    }
    __syncthreads();

    // final LayerNorm (same gamma/beta, per reference): 2 rows per wave
#pragma unroll
    for (int rr = 0; rr < 2; ++rr) {
        int row = wave * 2 + rr;
        float x[4], s = 0.f;
#pragma unroll
        for (int j = 0; j < 4; ++j) { x[j] = sOut[row * NIN + lane + 32 * j]; s += x[j]; }
        float mu = wave_sum(s) * (1.0f / NIN);
        float vs = 0.f;
#pragma unroll
        for (int j = 0; j < 4; ++j) { float dd = x[j] - mu; vs += dd * dd; }
        float rs = rsqrtf(wave_sum(vs) * (1.0f / NIN) + 1e-5f);
#pragma unroll
        for (int j = 0; j < 4; ++j) {
            int c = lane + 32 * j;
            out[(size_t)(rbase + row) * NIN + c] = (x[j] - mu) * rs * g[c] + bb[c];
        }
    }
}

// ---------------- host launch ----------------
extern "C" void kernel_launch(void* const* d_in, const int* in_sizes, int n_in,
                              void* d_out, int out_size, void* d_ws, size_t ws_size,
                              hipStream_t stream) {
    (void)n_in; (void)out_size; (void)ws_size;
    const float* feat = (const float*)d_in[0];
    const int*   src  = (const int*)d_in[1];
    const int*   dst  = (const int*)d_in[2];
    const float* Wq   = (const float*)d_in[3];
    const float* Wk   = (const float*)d_in[4];
    const float* Wv   = (const float*)d_in[5];
    const float* ln_g = (const float*)d_in[6];
    const float* ln_b = (const float*)d_in[7];
    const float* W1   = (const float*)d_in[8];
    const float* b1   = (const float*)d_in[9];
    const float* pa   = (const float*)d_in[10];
    const float* W2   = (const float*)d_in[11];
    const float* b2   = (const float*)d_in[12];
    float* out = (float*)d_out;

    const int N = in_sizes[0] / NIN;   // 50000 (multiple of 16)
    const int E = in_sizes[1];         // 640000

    char* ws = (char*)d_ws;
    size_t off = 0;
    auto alloc = [&](size_t bytes) -> void* {
        off = (off + 255) & ~(size_t)255;
        void* p = ws + off; off += bytes; return p;
    };
    _Float16* feat16 = (_Float16*)alloc((size_t)N * NIN * 2);
    _Float16* wqkvT  = (_Float16*)alloc((size_t)QKVC * NIN * 2);
    _Float16* w1T    = (_Float16*)alloc((size_t)DFF * NIN * 2);
    _Float16* w2T    = (_Float16*)alloc((size_t)NIN * DFF * 2);
    _Float16* qkv16  = (_Float16*)alloc((size_t)N * QKVC * 2);
    float*    escore = (float*)alloc((size_t)E * NHEAD * 4);
    unsigned* nmax   = (unsigned*)alloc((size_t)N * NHEAD * 4);
    float*    denom  = (float*)alloc((size_t)N * NHEAD * 4);
    float*    ft2    = (float*)alloc((size_t)N * NIN * 4);
    float*    rst32  = (float*)alloc((size_t)N * NIN * 4);
    _Float16* rst16  = (_Float16*)alloc((size_t)N * NIN * 2);

    const int TB = 256;
    // prep
    k_f32_to_f16<<<(N * NIN + TB - 1) / TB, TB, 0, stream>>>(feat, feat16, N * NIN);
    k_build_wqkvT<<<(QKVC * NIN + TB - 1) / TB, TB, 0, stream>>>(Wq, Wk, Wv, wqkvT);
    k_build_w1T<<<(DFF * NIN + TB - 1) / TB, TB, 0, stream>>>(W1, w1T);
    k_build_w2T<<<(NIN * DFF + TB - 1) / TB, TB, 0, stream>>>(W2, w2T);
    k_zero<<<(N * NHEAD + TB - 1) / TB, TB, 0, stream>>>((float*)nmax, N * NHEAD);
    k_zero<<<(N * NHEAD + TB - 1) / TB, TB, 0, stream>>>(denom, N * NHEAD);
    k_zero<<<(N * NIN + TB - 1) / TB, TB, 0, stream>>>(ft2, N * NIN);

    // QKV projection (WMMA)
    dim3 gq(N / 16, QKVC / (16 * 8));          // (3125, 3)
    k_qkv_gemm<<<gq, TB, 0, stream>>>(feat16, wqkvT, qkv16);

    // edge softmax + aggregation
    int eThreads = E * NHEAD;
    int eBlocks = (eThreads + TB - 1) / TB;
    k_edge_score<<<eBlocks, TB, 0, stream>>>(qkv16, src, dst, escore, nmax, E);
    k_edge_exp<<<eBlocks, TB, 0, stream>>>(dst, escore, nmax, denom, E);
    k_edge_aggregate<<<eBlocks, TB, 0, stream>>>(qkv16, src, dst, escore, denom, ft2, E);

    // residual + LN1
    k_residual_ln<<<(N + 7) / 8, TB, 0, stream>>>(ft2, feat, ln_g, ln_b, rst32, rst16, N);

    // fused FFN + final LN (WMMA + async-LDS + LDS)
    k_ffn_fused<<<N / 16, TB, 0, stream>>>(rst16, rst32, w1T, b1, pa, w2T, b2, ln_g, ln_b, out);
}
